// UOT_Head_20229295965010
// MI455X (gfx1250) — compile-verified, matrix operand order checked by reading, not run on hardware
//
#include <hip/hip_runtime.h>
#include <math.h>

// Problem constants (fixed by reference setup_inputs)
constexpr int Bn = 32;
constexpr int Dd = 512;
constexpr int Mm = 1024;
constexpr int Nn = 1024;

typedef __bf16 bf16_t;
typedef bf16_t v16bf __attribute__((ext_vector_type(16)));
typedef float  v8f   __attribute__((ext_vector_type(8)));
typedef unsigned int uint4v __attribute__((ext_vector_type(4)));
typedef int v4i32 __attribute__((vector_size(4 * sizeof(int))));  // matches builtin pointee

struct Frag2 { uint4v lo, hi; };

#if defined(__AMDGCN__) && __has_builtin(__builtin_amdgcn_global_load_async_to_lds_b128)
#define HAVE_ASYNC 1
#else
#define HAVE_ASYNC 0
#endif

// generic -> AS1 (global) / AS3 (LDS offset = low 32 bits of flat LDS address),
// typed as v4i32* to match the async builtin's parameter types.
#define GLB_AS(p) ((__attribute__((address_space(1))) v4i32*)(unsigned long long)(const void*)(p))
#define LDS_AS(p) ((__attribute__((address_space(3))) v4i32*)(unsigned int)(unsigned long long)(const void*)(p))

__device__ inline void wait_async0() {
#if HAVE_ASYNC
#if __has_builtin(__builtin_amdgcn_s_wait_asynccnt)
  __builtin_amdgcn_s_wait_asynccnt(0);
#else
  asm volatile("s_wait_asynccnt 0" ::: "memory");
#endif
#endif
}

// Assemble a 16-element bf16 WMMA fragment from two 16B-aligned LDS chunks.
// LDS tile layout: [row][k], row stride = 64 bf16 (128 bytes).
// Per ISA 16-bit A/B layout: lane<16 holds K {0..7} and {16..23};
// lane>=16 holds K {8..15} and {24..31} (koff = 0 or 8, plus 32 for 2nd K-substep).
__device__ inline v16bf load_frag64(const unsigned short* lds, int row, int ko) {
  const uint4v* p0 = reinterpret_cast<const uint4v*>(lds + row * 64 + ko);
  const uint4v* p1 = reinterpret_cast<const uint4v*>(lds + row * 64 + ko + 16);
  Frag2 f; f.lo = *p0; f.hi = *p1;
  return __builtin_bit_cast(v16bf, f);
}

__device__ inline unsigned short f2bf(float x) {  // fp32 -> bf16 RNE
  unsigned int u = __float_as_uint(x);
  u += 0x7FFFu + ((u >> 16) & 1u);
  return (unsigned short)(u >> 16);
}

// ---------------- transpose+convert: (B,D,M) fp32 -> (B,M,D) bf16 ----------------
__global__ void convert_tr(const float* __restrict__ in, unsigned short* __restrict__ outp) {
  __shared__ unsigned short t[32][33];
  int b  = blockIdx.z;
  int m0 = blockIdx.x * 32, d0 = blockIdx.y * 32;
  int tid = threadIdx.x;  // 256
  #pragma unroll
  for (int i = 0; i < 4; ++i) {
    int idx = tid + 256 * i;   // 0..1023
    int d = idx >> 5, m = idx & 31;
    float x = in[((size_t)b * Dd + d0 + d) * Mm + m0 + m];
    t[m][d] = f2bf(x);
  }
  __syncthreads();
  #pragma unroll
  for (int i = 0; i < 4; ++i) {
    int idx = tid + 256 * i;
    int m = idx >> 5, d = idx & 31;
    outp[((size_t)b * Mm + m0 + m) * Dd + d0 + d] = t[m][d];
  }
}

// ---------------- norms: out[b*M+m] = ||emb[b,:,m]|| (fp32 source) ----------------
__global__ void norm_kernel(const float* __restrict__ emb, float* __restrict__ out) {
  int idx = blockIdx.x * 256 + threadIdx.x;  // b*M + m
  int b = idx >> 10;
  int m = idx & 1023;
  const float* p = emb + (size_t)b * Dd * Mm + m;
  float s = 0.f;
  #pragma unroll 8
  for (int d = 0; d < Dd; ++d) { float x = p[(size_t)d * Mm]; s += x * x; }
  out[idx] = sqrtf(s);
}

// ---------------- fused GEMM + cosine + exp(-C/l) -> K ----------------
// grid (M/64, N/64, B), block 128 (4 waves). Each wave: 2x2 WMMA 16x16 tiles.
// bf16 inputs row-major [b][m][d]; async double-buffered LDS staging, K-step 64.
__device__ inline void stage_tile(unsigned short* ldsA, unsigned short* ldsB,
                                  const unsigned short* Ab, const unsigned short* Bb,
                                  int k0, int tid) {
  #pragma unroll
  for (int r = 0; r < 4; ++r) {
    int idx = tid + 128 * r;  // 0..511
    int row = idx >> 3;       // 0..63
    int c   = (idx & 7) * 8;  // bf16 elem offset within 64-wide row, 16B chunks
    const unsigned short* ga = Ab + (size_t)row * Dd + k0 + c;
    const unsigned short* gb = Bb + (size_t)row * Dd + k0 + c;
    unsigned short* la = ldsA + row * 64 + c;
    unsigned short* lb = ldsB + row * 64 + c;
#if HAVE_ASYNC
    __builtin_amdgcn_global_load_async_to_lds_b128(GLB_AS(ga), LDS_AS(la), 0, 0);
    __builtin_amdgcn_global_load_async_to_lds_b128(GLB_AS(gb), LDS_AS(lb), 0, 0);
#else
    *reinterpret_cast<uint4v*>(la) = *reinterpret_cast<const uint4v*>(ga);
    *reinterpret_cast<uint4v*>(lb) = *reinterpret_cast<const uint4v*>(gb);
#endif
  }
}

__global__ __launch_bounds__(128)
void gemm_expK(const unsigned short* __restrict__ xa, const unsigned short* __restrict__ xb,
               const float* __restrict__ n1, const float* __restrict__ n2,
               const float* __restrict__ lp, float* __restrict__ Kmat) {
  __shared__ unsigned short lA[2][64 * 64];
  __shared__ unsigned short lB[2][64 * 64];
  int tid  = threadIdx.x;
  int lane = tid & 31, wave = tid >> 5;
  int wm = wave & 1, wn = wave >> 1;
  int b  = blockIdx.z;
  int m0 = blockIdx.x * 64, n0 = blockIdx.y * 64;
  const unsigned short* Ab = xa + (size_t)(b * Mm + m0) * Dd;
  const unsigned short* Bb = xb + (size_t)(b * Nn + n0) * Dd;

  v8f acc[2][2] = {};
  int koff = (lane < 16) ? 0 : 8;
  int lrow = lane & 15;

  stage_tile(lA[0], lB[0], Ab, Bb, 0, tid);
  int cur = 0;
  for (int k0 = 0; k0 < Dd; k0 += 64) {
    wait_async0();
    __syncthreads();
    if (k0 + 64 < Dd) stage_tile(lA[cur ^ 1], lB[cur ^ 1], Ab, Bb, k0 + 64, tid);

    const unsigned short* cA = lA[cur];
    const unsigned short* cB = lB[cur];
    #pragma unroll
    for (int kk = 0; kk < 2; ++kk) {
      int ko = kk * 32 + koff;
      v16bf af[2], bfv[2];
      #pragma unroll
      for (int ti = 0; ti < 2; ++ti) af[ti]  = load_frag64(cA, wm * 32 + ti * 16 + lrow, ko);
      #pragma unroll
      for (int tj = 0; tj < 2; ++tj) bfv[tj] = load_frag64(cB, wn * 32 + tj * 16 + lrow, ko);
      #pragma unroll
      for (int ti = 0; ti < 2; ++ti)
        #pragma unroll
        for (int tj = 0; tj < 2; ++tj)
          acc[ti][tj] = __builtin_amdgcn_wmma_f32_16x16x32_bf16(
              false, af[ti], false, bfv[tj], (short)0, acc[ti][tj], false, false);
    }
    cur ^= 1;
  }

  float inv_l = 1.0f / fmaxf(lp[0], 1e-8f);
  int mofs = (lane < 16) ? 0 : 8;
  #pragma unroll
  for (int ti = 0; ti < 2; ++ti)
    #pragma unroll
    for (int tj = 0; tj < 2; ++tj)
      #pragma unroll
      for (int r = 0; r < 8; ++r) {
        int m = m0 + wm * 32 + ti * 16 + r + mofs;
        int n = n0 + wn * 32 + tj * 16 + lrow;
        float S  = acc[ti][tj][r];
        float rd = __builtin_amdgcn_rcpf(fmaxf(n1[b * Mm + m] * n2[b * Nn + n], 1e-6f));
        float Cc = 1.0f - S * rd;
        Kmat[(size_t)(b * Mm + m) * Nn + n] = __expf(-Cc * inv_l);
      }
}

// ---------------- Sinkhorn ----------------
__global__ void init_v(float* __restrict__ v) {
  v[blockIdx.x * 256 + threadIdx.x] = 1.0f / Nn;
}

// wave-per-row: u[b,m] = (a / clamp(sum_n K*v))^fi
__global__ void row_update(const float* __restrict__ Kmat, const float* __restrict__ v,
                           float* __restrict__ u, const float* __restrict__ lp,
                           const float* __restrict__ rp) {
  int row  = blockIdx.x * 8 + (threadIdx.x >> 5);
  int lane = threadIdx.x & 31;
  int b    = row >> 10;
  const float* Kr = Kmat + (size_t)row * Nn;
  const float* vb = v + b * Nn;
  float s = 0.f;
  for (int n = lane; n < Nn; n += 32) s += Kr[n] * vb[n];
  #pragma unroll
  for (int off = 16; off; off >>= 1) s += __shfl_xor(s, off, 32);
  if (lane == 0) {
    float l = lp[0], rho = rp[0];
    float fi = rho / fmaxf(rho + l, 1e-8f);
    u[row] = __powf((1.0f / Mm) / fmaxf(s, 1e-8f), fi);
  }
}

// thread-per-column: v[b,n] = (b / clamp(sum_m K*u))^fi ; u staged in LDS
__global__ void col_update(const float* __restrict__ Kmat, const float* __restrict__ u,
                           float* __restrict__ v, const float* __restrict__ lp,
                           const float* __restrict__ rp) {
  __shared__ float su[Mm];
  int b = blockIdx.y, tid = threadIdx.x;
  for (int i = tid; i < Mm; i += 256) su[i] = u[b * Mm + i];
  __syncthreads();
  int n = blockIdx.x * 256 + tid;
  const float* Kb = Kmat + (size_t)b * Mm * Nn + n;
  float s = 0.f;
  for (int m = 0; m < Mm; ++m) s += Kb[(size_t)m * Nn] * su[m];
  float l = lp[0], rho = rp[0];
  float fi = rho / fmaxf(rho + l, 1e-8f);
  v[b * Nn + n] = __powf((1.0f / Nn) / fmaxf(s, 1e-8f), fi);
}

// ---------------- P = u*K*v + fused row sums & weighted_ref ----------------
__global__ void p_kernel(const float* __restrict__ Kmat, const float* __restrict__ u,
                         const float* __restrict__ v, const float* __restrict__ tgt,
                         float* __restrict__ P, float* __restrict__ rowP,
                         float* __restrict__ wref) {
  int row  = blockIdx.x * 8 + (threadIdx.x >> 5);
  int lane = threadIdx.x & 31;
  int b    = row >> 10;
  const float* Kr = Kmat + (size_t)row * Nn;
  const float* vb = v + b * Nn;
  const float* tx = tgt + (size_t)(b * 3 + 0) * Nn;
  const float* ty = tgt + (size_t)(b * 3 + 1) * Nn;
  const float* tz = tgt + (size_t)(b * 3 + 2) * Nn;
  float ur = u[row];
  float* Pr = P + (size_t)row * Nn;
  float sp = 0.f, sx = 0.f, sy = 0.f, sz = 0.f;
  for (int n = lane; n < Nn; n += 32) {
    float p = ur * Kr[n] * vb[n];
    Pr[n] = p;
    sp += p; sx += p * tx[n]; sy += p * ty[n]; sz += p * tz[n];
  }
  #pragma unroll
  for (int off = 16; off; off >>= 1) {
    sp += __shfl_xor(sp, off, 32);
    sx += __shfl_xor(sx, off, 32);
    sy += __shfl_xor(sy, off, 32);
    sz += __shfl_xor(sz, off, 32);
  }
  if (lane == 0) {
    rowP[row] = sp;
    float inv = 1.0f / (sp + 1e-6f);
    wref[(size_t)row * 3 + 0] = sx * inv;
    wref[(size_t)row * 3 + 1] = sy * inv;
    wref[(size_t)row * 3 + 2] = sz * inv;
  }
}

// ---------------- Kabsch + 3x3 SVD per batch ----------------
__device__ inline float blockReduceSum(float val, float* sdata) {
  int tid = threadIdx.x;
  sdata[tid] = val;
  __syncthreads();
  for (int s = 128; s > 0; s >>= 1) {
    if (tid < s) sdata[tid] += sdata[tid + s];
    __syncthreads();
  }
  float r = sdata[0];
  __syncthreads();
  return r;
}

__device__ void jacobi_eig3(float A[3][3], float V[3][3]) {
  #pragma unroll
  for (int i = 0; i < 3; ++i)
    #pragma unroll
    for (int j = 0; j < 3; ++j) V[i][j] = (i == j) ? 1.f : 0.f;
  for (int sweep = 0; sweep < 12; ++sweep) {
    int p = (sweep % 3 == 0) ? 0 : (sweep % 3 == 1 ? 0 : 1);
    int q = (sweep % 3 == 0) ? 1 : (sweep % 3 == 1 ? 2 : 2);
    float apq = A[p][q];
    if (fabsf(apq) < 1e-12f) continue;
    float theta = (A[q][q] - A[p][p]) / (2.f * apq);
    float t = copysignf(1.f, theta) / (fabsf(theta) + sqrtf(theta * theta + 1.f));
    float c = 1.f / sqrtf(t * t + 1.f);
    float s = t * c;
    for (int k = 0; k < 3; ++k) {
      float akp = A[k][p], akq = A[k][q];
      A[k][p] = c * akp - s * akq;
      A[k][q] = s * akp + c * akq;
    }
    for (int k = 0; k < 3; ++k) {
      float apk = A[p][k], aqk = A[q][k];
      A[p][k] = c * apk - s * aqk;
      A[q][k] = s * apk + c * aqk;
    }
    for (int k = 0; k < 3; ++k) {
      float vkp = V[k][p], vkq = V[k][q];
      V[k][p] = c * vkp - s * vkq;
      V[k][q] = s * vkp + c * vkq;
    }
  }
}

__global__ void kabsch_kernel(const float* __restrict__ src, const float* __restrict__ rowP,
                              const float* __restrict__ wref, float* __restrict__ Rout,
                              float* __restrict__ tout) {
  __shared__ float sdata[256];
  int b = blockIdx.x, tid = threadIdx.x;
  const float* rp = rowP + b * Mm;
  const float* wr = wref + (size_t)b * Mm * 3;
  const float* ax = src + (size_t)(b * 3 + 0) * Mm;
  const float* ay = src + (size_t)(b * 3 + 1) * Mm;
  const float* az = src + (size_t)(b * 3 + 2) * Mm;

  float loc = 0.f;
  for (int m = tid; m < Mm; m += 256) loc += rp[m];
  float total = blockReduceSum(loc, sdata);
  float invT = 1.0f / (total + 1e-6f);

  float ca[3] = {0, 0, 0}, cb[3] = {0, 0, 0};
  for (int m = tid; m < Mm; m += 256) {
    float w = rp[m] * invT;
    ca[0] += ax[m] * w; ca[1] += ay[m] * w; ca[2] += az[m] * w;
    cb[0] += wr[m * 3 + 0] * w; cb[1] += wr[m * 3 + 1] * w; cb[2] += wr[m * 3 + 2] * w;
  }
  float CA[3], CB[3];
  for (int c = 0; c < 3; ++c) CA[c] = blockReduceSum(ca[c], sdata);
  for (int c = 0; c < 3; ++c) CB[c] = blockReduceSum(cb[c], sdata);

  float cv[3][3] = {};
  for (int m = tid; m < Mm; m += 256) {
    float w  = rp[m] * invT;
    float dx = ax[m] - CA[0], dy = ay[m] - CA[1], dz = az[m] - CA[2];
    float bx = (wr[m * 3 + 0] - CB[0]) * w;
    float by = (wr[m * 3 + 1] - CB[1]) * w;
    float bz = (wr[m * 3 + 2] - CB[2]) * w;
    cv[0][0] += dx * bx; cv[0][1] += dx * by; cv[0][2] += dx * bz;
    cv[1][0] += dy * bx; cv[1][1] += dy * by; cv[1][2] += dy * bz;
    cv[2][0] += dz * bx; cv[2][1] += dz * by; cv[2][2] += dz * bz;
  }
  for (int i = 0; i < 3; ++i)
    for (int j = 0; j < 3; ++j) cv[i][j] = blockReduceSum(cv[i][j], sdata);

  if (tid == 0) {
    float S[3][3], V[3][3];
    for (int i = 0; i < 3; ++i)
      for (int j = 0; j < 3; ++j) {
        float s = 0.f;
        for (int k = 0; k < 3; ++k) s += cv[k][i] * cv[k][j];
        S[i][j] = s;
      }
    jacobi_eig3(S, V);
    float ev[3] = {S[0][0], S[1][1], S[2][2]};
    for (int i = 0; i < 2; ++i)
      for (int j = 0; j < 2 - i; ++j)
        if (ev[j] < ev[j + 1]) {
          float te = ev[j]; ev[j] = ev[j + 1]; ev[j + 1] = te;
          for (int k = 0; k < 3; ++k) { float tv = V[k][j]; V[k][j] = V[k][j + 1]; V[k][j + 1] = tv; }
        }
    float U[3][3];
    for (int i = 0; i < 3; ++i) {
      float ui[3];
      for (int k = 0; k < 3; ++k) {
        float s = 0.f;
        for (int j = 0; j < 3; ++j) s += cv[k][j] * V[j][i];
        ui[k] = s;
      }
      float nrm = sqrtf(ui[0] * ui[0] + ui[1] * ui[1] + ui[2] * ui[2]);
      if (nrm > 1e-10f) {
        U[0][i] = ui[0] / nrm; U[1][i] = ui[1] / nrm; U[2][i] = ui[2] / nrm;
      } else if (i == 2) {
        U[0][2] = U[1][0] * U[2][1] - U[2][0] * U[1][1];
        U[1][2] = U[2][0] * U[0][1] - U[0][0] * U[2][1];
        U[2][2] = U[0][0] * U[1][1] - U[1][0] * U[0][1];
      } else {
        U[0][i] = (i == 0); U[1][i] = (i == 1); U[2][i] = 0.f;
      }
    }
    float R[3][3];
    for (int i = 0; i < 3; ++i)
      for (int j = 0; j < 3; ++j) {
        float s = 0.f;
        for (int k = 0; k < 3; ++k) s += V[i][k] * U[j][k];
        R[i][j] = s;
      }
    float det = R[0][0] * (R[1][1] * R[2][2] - R[1][2] * R[2][1])
              - R[0][1] * (R[1][0] * R[2][2] - R[1][2] * R[2][0])
              + R[0][2] * (R[1][0] * R[2][1] - R[1][1] * R[2][0]);
    if (det <= 0.f) {
      for (int k = 0; k < 3; ++k) V[k][2] = -V[k][2];
      for (int i = 0; i < 3; ++i)
        for (int j = 0; j < 3; ++j) {
          float s = 0.f;
          for (int k = 0; k < 3; ++k) s += V[i][k] * U[j][k];
          R[i][j] = s;
        }
    }
    for (int i = 0; i < 3; ++i)
      for (int j = 0; j < 3; ++j) Rout[b * 9 + i * 3 + j] = R[i][j];
    for (int c = 0; c < 3; ++c) {
      float s = 0.f;
      for (int j = 0; j < 3; ++j) s += R[c][j] * CA[j];
      tout[b * 3 + c] = CB[c] - s;
    }
  }
}

extern "C" void kernel_launch(void* const* d_in, const int* in_sizes, int n_in,
                              void* d_out, int out_size, void* d_ws, size_t ws_size,
                              hipStream_t stream) {
  (void)in_sizes; (void)n_in; (void)out_size; (void)ws_size;
  const float* src_emb = (const float*)d_in[0];  // (B, D, M)
  const float* tgt_emb = (const float*)d_in[1];  // (B, D, N)
  const float* src     = (const float*)d_in[2];  // (B, 3, M)
  const float* tgt     = (const float*)d_in[3];  // (B, 3, N)
  const float* lp      = (const float*)d_in[4];  // scalar
  const float* rp      = (const float*)d_in[5];  // scalar

  float* out  = (float*)d_out;
  float* Rout = out;                    // B*9
  float* tout = out + Bn * 9;           // B*3
  float* Pout = out + Bn * 9 + Bn * 3;  // B*M*N

  // workspace layout
  float* Kmat = (float*)d_ws;                 // B*M*N
  float* nrm1 = Kmat + (size_t)Bn * Mm * Nn;  // B*M
  float* nrm2 = nrm1 + Bn * Mm;               // B*N
  float* u    = nrm2 + Bn * Nn;               // B*M
  float* v    = u + Bn * Mm;                  // B*N
  float* rowP = v + Bn * Nn;                  // B*M
  float* wref = rowP + Bn * Mm;               // B*M*3
  unsigned short* xa = (unsigned short*)(wref + (size_t)Bn * Mm * 3);  // B*M*D bf16
  unsigned short* xb = xa + (size_t)Bn * Mm * Dd;                      // B*N*D bf16

  convert_tr<<<dim3(Mm / 32, Dd / 32, Bn), 256, 0, stream>>>(src_emb, xa);
  convert_tr<<<dim3(Nn / 32, Dd / 32, Bn), 256, 0, stream>>>(tgt_emb, xb);

  norm_kernel<<<(Bn * Mm) / 256, 256, 0, stream>>>(src_emb, nrm1);
  norm_kernel<<<(Bn * Nn) / 256, 256, 0, stream>>>(tgt_emb, nrm2);

  gemm_expK<<<dim3(Mm / 64, Nn / 64, Bn), 128, 0, stream>>>(xa, xb, nrm1, nrm2, lp, Kmat);

  init_v<<<(Bn * Nn) / 256, 256, 0, stream>>>(v);
  for (int it = 0; it < 5; ++it) {
    row_update<<<(Bn * Mm) / 8, 256, 0, stream>>>(Kmat, v, u, lp, rp);
    col_update<<<dim3(Nn / 256, Bn), 256, 0, stream>>>(Kmat, u, v, lp, rp);
  }

  p_kernel<<<(Bn * Mm) / 8, 256, 0, stream>>>(Kmat, u, v, tgt, Pout, rowP, wref);
  kabsch_kernel<<<Bn, 256, 0, stream>>>(src, rowP, wref, Rout, tout);
}
